// KMEAN_Memory_25177098289570
// MI455X (gfx1250) — compile-verified, matrix operand order checked by reading, not run on hardware
//
#include <hip/hip_runtime.h>
#include <hip/hip_bf16.h>
#include <stdint.h>

// ---------------- problem constants (match reference) ----------------
#define NKC   64        // centroids
#define NPTS  1024      // samples (1024 sequences)
#define DDIM  98304     // 128*768 flattened feature dim
#define NITER 10

// ---------------- GEMM tiling ----------------
#define TILE_M  128     // rows per block (8 wave32 x 16)
#define TILE_D  64      // D-chunk staged to LDS per step
#define DSPLIT  1024    // D range owned by one block (partial dot, atomic reduce)
#define LPAD    (TILE_D / 2 + 3)   // 35 dwords/row: odd+coprime stride -> conflict-free frags

typedef __attribute__((ext_vector_type(16))) __bf16 v16bf;
typedef __attribute__((ext_vector_type(8)))  float  v8f;

// round-to-nearest-even fp32 -> bf16 bits
__device__ __forceinline__ unsigned f2bf(float f) {
    unsigned u = __float_as_uint(f);
    return (u + 0x7FFFu + ((u >> 16) & 1u)) >> 16;
}
__device__ __forceinline__ float bf2f(unsigned hb) { return __uint_as_float(hb << 16); }

// split two fp32 into packed bf16 hi-pair and lo-pair (x = hi + lo)
__device__ __forceinline__ void split2(float x0, float x1, unsigned& hi, unsigned& lo) {
    unsigned h0 = f2bf(x0), h1 = f2bf(x1);
    float    r0 = x0 - bf2f(h0), r1 = x1 - bf2f(h1);
    hi = h0 | (f2bf(0.0f), h1 << 16);
    lo = f2bf(r0) | (f2bf(r1) << 16);
}

// ---------------------------------------------------------------------
// dot[n][k] += X[n,:] . C[k,:] over this block's D-range, via split-bf16 WMMA
// grid = (NPTS/TILE_M, DDIM/DSPLIT), block = 256
// ---------------------------------------------------------------------
__global__ __launch_bounds__(256)
void gemm_wmma(const float* __restrict__ X, const float* __restrict__ C,
               float* __restrict__ dotNK) {
    __shared__ unsigned lxh[TILE_M * LPAD];
    __shared__ unsigned lxl[TILE_M * LPAD];
    __shared__ unsigned lch[NKC * LPAD];
    __shared__ unsigned lcl[NKC * LPAD];

    const int tid  = threadIdx.x;
    const int lane = tid & 31;
    const int wave = tid >> 5;
    const int half = lane >> 4;    // K-half select per 16-bit A/B VGPR layout
    const int l15  = lane & 15;
    const int m0   = blockIdx.x * TILE_M;
    const int d0   = blockIdx.y * DSPLIT;

    v8f acc[4];
#pragma unroll
    for (int t = 0; t < 4; ++t)
#pragma unroll
        for (int e = 0; e < 8; ++e) acc[t][e] = 0.0f;

    for (int dc = 0; dc < DSPLIT; dc += TILE_D) {
        const int d = d0 + dc;

        // ---- stage X tile (128 x 64 fp32) as bf16 hi/lo pairs ----
#pragma unroll
        for (int i = tid; i < TILE_M * (TILE_D / 4); i += 256) {
            const int r  = i >> 4;        // 16 float4 per row
            const int c4 = i & 15;
            const float4 v = *(const float4*)(X + (size_t)(m0 + r) * DDIM + d + c4 * 4);
            unsigned h0, l0, h1, l1;
            split2(v.x, v.y, h0, l0);
            split2(v.z, v.w, h1, l1);
            lxh[r * LPAD + c4 * 2]     = h0;  lxl[r * LPAD + c4 * 2]     = l0;
            lxh[r * LPAD + c4 * 2 + 1] = h1;  lxl[r * LPAD + c4 * 2 + 1] = l1;
        }
        // ---- stage C tile (64 x 64 fp32) ----
#pragma unroll
        for (int i = tid; i < NKC * (TILE_D / 4); i += 256) {
            const int r  = i >> 4;
            const int c4 = i & 15;
            const float4 v = *(const float4*)(C + (size_t)r * DDIM + d + c4 * 4);
            unsigned h0, l0, h1, l1;
            split2(v.x, v.y, h0, l0);
            split2(v.z, v.w, h1, l1);
            lch[r * LPAD + c4 * 2]     = h0;  lcl[r * LPAD + c4 * 2]     = l0;
            lch[r * LPAD + c4 * 2 + 1] = h1;  lcl[r * LPAD + c4 * 2 + 1] = l1;
        }
        // prefetch next chunk of X while we compute (global_prefetch_b8)
        if (dc + TILE_D < DSPLIT)
            __builtin_prefetch(X + (size_t)(m0 + (tid >> 1)) * DDIM + d + TILE_D + ((tid & 1) << 5), 0, 0);
        __syncthreads();

        // ---- 2 k-steps of K=32, 4 centroid tiles, 3 WMMAs each (hi*hi + hi*lo + lo*hi) ----
#pragma unroll
        for (int ks = 0; ks < 2; ++ks) {
            const int kb = ks * 16;   // dword offset of this K=32 group
            union { v16bf v; unsigned u[8]; } Ah, Al;
#pragma unroll
            for (int p = 0; p < 8; ++p) {
                const int idx = kb + (p & 3) + ((p >> 2) << 3) + half * 4;
                const int row = wave * 16 + l15;
                Ah.u[p] = lxh[row * LPAD + idx];
                Al.u[p] = lxl[row * LPAD + idx];
            }
#pragma unroll
            for (int t = 0; t < 4; ++t) {
                union { v16bf v; unsigned u[8]; } Bh, Bl;
#pragma unroll
                for (int p = 0; p < 8; ++p) {
                    const int idx = kb + (p & 3) + ((p >> 2) << 3) + half * 4;
                    const int row = t * 16 + l15;
                    Bh.u[p] = lch[row * LPAD + idx];
                    Bl.u[p] = lcl[row * LPAD + idx];
                }
                acc[t] = __builtin_amdgcn_wmma_f32_16x16x32_bf16(
                    false, Ah.v, false, Bh.v, (short)0, acc[t], false, false);
                acc[t] = __builtin_amdgcn_wmma_f32_16x16x32_bf16(
                    false, Ah.v, false, Bl.v, (short)0, acc[t], false, false);
                acc[t] = __builtin_amdgcn_wmma_f32_16x16x32_bf16(
                    false, Al.v, false, Bh.v, (short)0, acc[t], false, false);
            }
        }
        __syncthreads();
    }

    // ---- epilogue: C/D layout — VGPR e: M = e + 8*half, N = l15 (+16 per tile) ----
#pragma unroll
    for (int t = 0; t < 4; ++t)
#pragma unroll
        for (int e = 0; e < 8; ++e) {
            const int M  = e + half * 8;
            const int Nc = t * 16 + l15;
            atomicAdd(dotNK + (size_t)(m0 + wave * 16 + M) * NKC + Nc, acc[t][e]);
        }
}

// ---------------- row squared-norms: one block per row ----------------
__global__ __launch_bounds__(256)
void rownorm(const float* __restrict__ A, float* __restrict__ out) {
    const int row = blockIdx.x;
    const float4* p = (const float4*)(A + (size_t)row * DDIM);
    float s = 0.0f;
    for (int i = threadIdx.x; i < DDIM / 4; i += 256) {
        const float4 v = p[i];
        s += v.x * v.x + v.y * v.y + v.z * v.z + v.w * v.w;
    }
    __shared__ float red[256];
    red[threadIdx.x] = s;
    __syncthreads();
    for (int off = 128; off > 0; off >>= 1) {
        if (threadIdx.x < off) red[threadIdx.x] += red[threadIdx.x + off];
        __syncthreads();
    }
    if (threadIdx.x == 0) out[row] = red[0];
}

// init: choice_points[k] = argmin_n (x2[n] - 2*dot[n][k])    (sqrt/c2 monotone-invariant)
__global__ void argmin_per_centroid(const float* __restrict__ dotNK,
                                    const float* __restrict__ x2,
                                    int* __restrict__ cp) {
    const int k = threadIdx.x;
    if (k >= NKC) return;
    float best = 3.4e38f; int bi = 0;
    for (int n = 0; n < NPTS; ++n) {
        const float v = x2[n] - 2.0f * dotNK[n * NKC + k];
        if (v < best) { best = v; bi = n; }
    }
    cp[k] = bi;
}

// state0[k,:] = X[cp[k],:]   grid=(DDIM/256, NKC)
__global__ __launch_bounds__(256)
void gather_rows(const float* __restrict__ X, const int* __restrict__ cp,
                 float* __restrict__ st) {
    const int k   = blockIdx.y;
    const int col = blockIdx.x * 256 + threadIdx.x;
    st[(size_t)k * DDIM + col] = X[(size_t)cp[k] * DDIM + col];
}

// choice[n] = argmin_k (c2[k] - 2*dot[n][k]); histogram counts
__global__ __launch_bounds__(256)
void assign_kernel(const float* __restrict__ dotNK, const float* __restrict__ c2,
                   int* __restrict__ choice, int* __restrict__ counts) {
    const int n = blockIdx.x * 256 + threadIdx.x;
    if (n >= NPTS) return;
    float best = 3.4e38f; int bi = 0;
    for (int k = 0; k < NKC; ++k) {
        const float v = c2[k] - 2.0f * dotNK[n * NKC + k];
        if (v < best) { best = v; bi = k; }
    }
    choice[n] = bi;
    atomicAdd(counts + bi, 1);
}

// segment-sum + mean. Each thread owns one D-column exclusively across all 64
// clusters (sums[c*256+tid]) -> no atomics needed on the hot path.
// grid = DDIM/256
__global__ __launch_bounds__(256)
void update_kernel(const float* __restrict__ X, const int* __restrict__ choice,
                   const int* __restrict__ counts, const float* __restrict__ stcur,
                   float* __restrict__ stnext) {
    __shared__ float sums[NKC * 256];   // 64 KB
    const int tid = threadIdx.x;
    const int d0  = blockIdx.x * 256;
    for (int i = tid; i < NKC * 256; i += 256) sums[i] = 0.0f;   // zeros own columns only
    for (int n = 0; n < NPTS; ++n) {
        const int c = choice[n];
        sums[c * 256 + tid] += X[(size_t)n * DDIM + d0 + tid];
    }
    for (int k = 0; k < NKC; ++k) {
        const int cnt = counts[k];
        const float v = (cnt > 0) ? (sums[k * 256 + tid] / (float)cnt)
                                  : stcur[(size_t)k * DDIM + d0 + tid];
        stnext[(size_t)k * DDIM + d0 + tid] = v;
    }
}

__global__ void zero_f32(float* __restrict__ p, int n) {
    const int i = blockIdx.x * 256 + threadIdx.x;
    if (i < n) p[i] = 0.0f;
}
__global__ void zero_i32(int* __restrict__ p, int n) {
    const int i = blockIdx.x * 256 + threadIdx.x;
    if (i < n) p[i] = 0;
}
__global__ void choice_to_out(const int* __restrict__ ch, float* __restrict__ out) {
    const int n = blockIdx.x * 256 + threadIdx.x;
    if (n < NPTS) out[n] = (float)ch[n];
}

// ---------------------------------------------------------------------
extern "C" void kernel_launch(void* const* d_in, const int* in_sizes, int n_in,
                              void* d_out, int out_size, void* d_ws, size_t ws_size,
                              hipStream_t stream) {
    const float* X        = (const float*)d_in[0];   // [1024,128,768] -> [N,D]
    const float* centroid = (const float*)d_in[1];   // [64,D]

    float* outState  = (float*)d_out;                         // [64,D]
    float* outChoice = outState + (size_t)NKC * DDIM;         // [1024] as float

    // workspace layout (25.4 MB)
    float* dotNK  = (float*)d_ws;            // NPTS*NKC
    float* x2     = dotNK + NPTS * NKC;      // NPTS
    float* c2     = x2 + NPTS;               // NKC
    int*   counts = (int*)(c2 + NKC);        // NKC
    int*   choice = counts + NKC;            // NPTS
    int*   cp     = choice + NPTS;           // NKC
    float* stAlt  = (float*)(cp + NKC);      // NKC*DDIM (16B-aligned offset)

    const dim3 gGemm(NPTS / TILE_M, DDIM / DSPLIT);   // (8, 96)

    // ---- init: snap each centroid to its nearest sample ----
    rownorm<<<NPTS, 256, 0, stream>>>(X, x2);
    zero_f32<<<(NPTS * NKC + 255) / 256, 256, 0, stream>>>(dotNK, NPTS * NKC);
    gemm_wmma<<<gGemm, 256, 0, stream>>>(X, centroid, dotNK);
    argmin_per_centroid<<<1, 64, 0, stream>>>(dotNK, x2, cp);
    gather_rows<<<dim3(DDIM / 256, NKC), 256, 0, stream>>>(X, cp, outState);

    // ---- 10 Lloyd iterations, ping-pong so the final state lands in d_out ----
    float* cur = outState;
    float* nxt = stAlt;
    for (int it = 0; it < NITER; ++it) {
        rownorm<<<NKC, 256, 0, stream>>>(cur, c2);
        zero_f32<<<(NPTS * NKC + 255) / 256, 256, 0, stream>>>(dotNK, NPTS * NKC);
        zero_i32<<<1, 64, 0, stream>>>(counts, NKC);
        gemm_wmma<<<gGemm, 256, 0, stream>>>(X, cur, dotNK);
        assign_kernel<<<NPTS / 256, 256, 0, stream>>>(dotNK, c2, choice, counts);
        update_kernel<<<DDIM / 256, 256, 0, stream>>>(X, choice, counts, cur, nxt);
        float* t = cur; cur = nxt; nxt = t;
    }
    // after an even number of swaps, cur == outState

    choice_to_out<<<NPTS / 256, 256, 0, stream>>>(choice, outChoice);
}
